// DGCN_model_29454885716509
// MI455X (gfx1250) — matrix operand order repaired and checked by reading, compiled, not run on hardware
//
#include <hip/hip_runtime.h>
#include <hip/hip_bf16.h>

// ---------------------------------------------------------------------------
// Types for CDNA5 WMMA (wave32): D(16x16 f32) = A(16x32 bf16) * B(32x16 bf16) + C
// ---------------------------------------------------------------------------
typedef __attribute__((ext_vector_type(16))) __bf16 v16bf;
typedef __attribute__((ext_vector_type(8)))  __bf16 v8bf;
typedef __attribute__((ext_vector_type(8)))  float  v8f;
typedef __attribute__((ext_vector_type(4)))  unsigned int v4u;
typedef __attribute__((ext_vector_type(8)))  int v8i;
typedef __attribute__((ext_vector_type(4)))  int v4i;

#if __has_builtin(__builtin_amdgcn_tensor_load_to_lds)
#define USE_TDM 1
#else
#define USE_TDM 0
#endif

__device__ __forceinline__ unsigned short f2bf(float f) {
  union { float f; unsigned int u; } v; v.f = f;
  unsigned int u = v.u;
  u += 0x7FFFu + ((u >> 16) & 1u);   // round-to-nearest-even
  return (unsigned short)(u >> 16);
}

#if USE_TDM
// ---------------------------------------------------------------------------
// TDM: DMA a [rows x 64] bf16 tile (row pitch strideElems) from global into LDS.
// LDS layout: 144B row pitch (128B data + 16B hardware pad -> bank-friendly,
// rows land 36 dwords apart). Rows >= `rows` are zero-filled by TDM OOB
// semantics (tensor_dim1 = rows). D# per CDNA5 ISA ch.8.
// 6-arg builtin on this toolchain: (v4u, v8i, v4i, v4i, v8i, cpol).
// ---------------------------------------------------------------------------
__device__ __forceinline__ void tdmLoadTile(const unsigned short* gptr, unsigned ldsOff,
                                            int strideElems, int rows) {
  unsigned long long ga = (unsigned long long)(size_t)gptr;
  v4u g0 = { 1u,                                   // count=1, user descriptor
             ldsOff,                               // lds_addr (bytes)
             (unsigned)ga,                         // global_addr[31:0]
             ((unsigned)(ga >> 32) & 0x01FFFFFFu) | (2u << 30) };  // addr[56:32] | type=2
  v8i g1 = {
    (int)((1u << 16)                 // data_size = 2 bytes
          | (1u << 20)               // pad_enable
          | (4u << 22)               // pad_interval: every 32 dwords (128B = one row)
          | (3u << 25)),             // pad_amount: 4 dwords (16B) -> 144B row pitch
    (int)(64u << 16),                // tensor_dim0 = 64 elements
    (int)(((unsigned)rows) << 16),   // tensor_dim1 = valid rows (OOB rows -> 0)
    (int)(64u << 16),                // tile_dim0 = 64
    16,                              // tile_dim1 = 16, tile_dim2 = 0
    strideElems,                     // tensor_dim0_stride[31:0]
    0, 0 };                          // stride hi / dim1_stride unused (2D)
  v4i z4 = { 0, 0, 0, 0 };
  v8i z8 = { 0, 0, 0, 0, 0, 0, 0, 0 };
  __builtin_amdgcn_tensor_load_to_lds(g0, g1, z4, z4, z8, 0);
}
#endif

// ---------------------------------------------------------------------------
// Small utility kernels
// ---------------------------------------------------------------------------
__global__ void kFillOnes(float* p, int n) {
  int i = blockIdx.x * blockDim.x + threadIdx.x;
  if (i < n) p[i] = 1.0f;
}

__global__ void kZero(float* p, long n) {
  long i = (long)blockIdx.x * blockDim.x + threadIdx.x;
  long stride = (long)gridDim.x * blockDim.x;
  for (; i < n; i += stride) p[i] = 0.0f;
}

__global__ void kDegAccum(const int* __restrict__ dst, const float* __restrict__ w,
                          float* __restrict__ deg, int E) {
  int e = blockIdx.x * blockDim.x + threadIdx.x;
  if (e < E) atomicAdd(&deg[dst[e]], w ? w[e] : 1.0f);
}

__global__ void kRsqrt(float* p, int n) {
  int i = blockIdx.x * blockDim.x + threadIdx.x;
  if (i < n) p[i] = rsqrtf(p[i]);   // deg >= 1 (self loop), always > 0
}

__global__ void kConvertBf(const float* __restrict__ in, unsigned short* __restrict__ out, long n) {
  long i = (long)blockIdx.x * blockDim.x + threadIdx.x;
  if (i < n) out[i] = f2bf(in[i]);
}

// Wt[m, k] = bf16(W[k, m]);  W is [K, M] row-major, Wt is [M, K] row-major
__global__ void kTransposeConvertW(const float* __restrict__ W, unsigned short* __restrict__ Wt,
                                   int K, int M) {
  int idx = blockIdx.x * blockDim.x + threadIdx.x;
  if (idx >= K * M) return;
  int mo = idx / K;
  int k  = idx - mo * K;
  Wt[idx] = f2bf(W[(long)k * M + mo]);
}

// ---------------------------------------------------------------------------
// WMMA bf16 GEMM with TDM-staged A tiles:
//   C[Nrows, M] = A[Nrows, K] (bf16) x Bt[M, K]^T (bf16)
// One block (8 waves) computes a 16 x 128 output slab. Per K-step, wave 0
// DMAs a shared 16x64 A chunk into LDS via tensor_load_to_lds (double
// buffered, s_wait_tensorcnt), and each wave runs TWO 16x16x32 WMMAs per
// barrier pair (amortizes sync; halves TDM issue rate vs 32-wide chunks).
//
// A fragment (16x32 bf16): lane = m + 16*half; elems 0..7 -> K = half*8 + j,
//                          elems 8..15 -> K = 16 + half*8 + j   (two b128 reads)
// B fragment (32x16 bf16): lane = n + 16*half; elems 0..7 -> K = half*8 + j, 8..15 -> K = 16 + half*8 + j
// D (16x16 f32): VGPR r: lanes 0-15 -> M=r, lanes 16-31 -> M=r+8, N = lane%16
// ---------------------------------------------------------------------------
#define LDS_ROW_SHORTS 72   // 64 data + 8 pad shorts = 144B pitch (matches TDM pad)
#define LDS_BUF_BYTES  (16 * LDS_ROW_SHORTS * 2)

__global__ void kGemmWmmaBf16(const unsigned short* __restrict__ A,
                              const unsigned short* __restrict__ Bt,
                              float* __restrict__ C,
                              int Nrows, int K, int M) {
  __shared__ unsigned short Abuf[2][16 * LDS_ROW_SHORTS];

  const int colGroups = M >> 7;                    // M / 128
  const int bm   = blockIdx.x / colGroups;
  const int bg   = blockIdx.x - bm * colGroups;
  const int wave = threadIdx.x >> 5;
  const int lane = threadIdx.x & 31;
  const int l16  = lane & 15;
  const int half = lane >> 4;
  const int rowBase = bm * 16;
  const int rowsValid = (Nrows - rowBase < 16) ? (Nrows - rowBase) : 16;
  const int col = bg * 128 + wave * 16 + l16;
  const int nsteps = K >> 6;                       // 64-wide K chunks

  const __bf16* bp = (const __bf16*)Bt + (long)col * K + half * 8;
  unsigned ldsBase = (unsigned)(size_t)&Abuf[0][0];

#if USE_TDM
  if (wave == 0) {
    tdmLoadTile(A + (long)rowBase * K, ldsBase, K, rowsValid);   // prefetch chunk 0
  }
#endif

  v8f c = {};
  for (int s = 0; s < nsteps; ++s) {
    const int buf = s & 1;
#if USE_TDM
    if (wave == 0) {
      if (s + 1 < nsteps) {   // prefetch next 16x64 chunk into the other buffer
        tdmLoadTile(A + (long)rowBase * K + (s + 1) * 64,
                    ldsBase + (unsigned)((buf ^ 1) * LDS_BUF_BYTES),
                    K, rowsValid);
        __builtin_amdgcn_s_wait_tensorcnt(1);   // chunk s complete (TDM in-order)
      } else {
        __builtin_amdgcn_s_wait_tensorcnt(0);
      }
    }
#else
    if (wave == 0) {          // cooperative fallback copy: lane -> (row, 32-short half-row)
      int r = l16;
      v8bf z = {};
      const v8bf* gp = (const v8bf*)((const __bf16*)A + (long)(rowBase + r) * K + s * 64 + half * 32);
      v8bf* lp = (v8bf*)&Abuf[buf][r * LDS_ROW_SHORTS + half * 32];
      #pragma unroll
      for (int q = 0; q < 4; ++q) lp[q] = (r < rowsValid) ? gp[q] : z;
    }
#endif
    __syncthreads();          // A chunk s visible to all waves

    #pragma unroll
    for (int ks = 0; ks < 2; ++ks) {               // two 16x16x32 WMMAs per barrier pair
      const __bf16* ap = (const __bf16*)&Abuf[buf][l16 * LDS_ROW_SHORTS + ks * 32 + half * 8];
      const __bf16* bq = bp + s * 64 + ks * 32;
      v16bf a, b;
      ((v8bf*)&a)[0] = *(const v8bf*)(ap);          // K = half*8 + 0..7
      ((v8bf*)&a)[1] = *(const v8bf*)(ap + 16);     // K = 16 + half*8 + 0..7
      ((v8bf*)&b)[0] = *(const v8bf*)(bq);          // K = half*8 + 0..7
      ((v8bf*)&b)[1] = *(const v8bf*)(bq + 16);     // K = 16 + half*8 + 0..7
      c = __builtin_amdgcn_wmma_f32_16x16x32_bf16(
          /*neg_a=*/false, a, /*neg_b=*/false, b,
          /*c_mod=*/(short)0, c, /*reuse_a=*/false, /*reuse_b=*/false);
    }

    __syncthreads();          // all waves done reading buf before it is re-filled
  }

  union { v8f v; float f[8]; } cu; cu.v = c;
  #pragma unroll
  for (int r = 0; r < 8; ++r) {
    int row = rowBase + half * 8 + r;
    if (row < Nrows) C[(long)row * M + col] = cu.f[r];
  }
}

// ---------------------------------------------------------------------------
// Edge-parallel scatter: out[dst, coff + f] += dinv[src]*w*dinv[dst] * h[src, f]
// out row width CW = 3*M (concat buffer). One feature per lane -> coalesced atomics.
// ---------------------------------------------------------------------------
__global__ void kScatter(const int* __restrict__ src, const int* __restrict__ dst,
                         const float* __restrict__ w, const float* __restrict__ dinv,
                         const float* __restrict__ h, float* __restrict__ out,
                         int E, int M, int CW, int coff, int edgesPerBlock) {
  int e0 = blockIdx.x * edgesPerBlock;
  int e1 = e0 + edgesPerBlock; if (e1 > E) e1 = E;
  for (int e = e0; e < e1; ++e) {
    int s = src[e], d = dst[e];
    float ww = w ? w[e] : 1.0f;
    float norm = dinv[s] * ww * dinv[d];
    const float* hrow = h + (long)s * M;
    float* orow = out + (long)d * CW + coff;
    for (int f = threadIdx.x; f < M; f += blockDim.x)
      atomicAdd(&orow[f], norm * hrow[f]);
  }
}

// ---------------------------------------------------------------------------
// Epilogue: add self-loop term dinv_k[n]^2 * h, add bias, ReLU.
// Writes fp32 result (final output on the last layer) and bf16 (next GEMM input).
// ---------------------------------------------------------------------------
__global__ void kEpilogue(const float* __restrict__ xcat_in, float* __restrict__ xout,
                          unsigned short* __restrict__ xbf, const float* __restrict__ h,
                          const float* __restrict__ dinvA, const float* __restrict__ dinvB,
                          const float* __restrict__ dinvC, const float* __restrict__ b,
                          int N, int M) {
  int idx = blockIdx.x * blockDim.x + threadIdx.x;
  if (idx >= N * M) return;
  int n = idx / M;
  int f = idx - n * M;
  float hv = h[idx];
  float bv = b[f];
  int CW = 3 * M;
  const float* dv[3] = { dinvA, dinvB, dinvC };
  #pragma unroll
  for (int k = 0; k < 3; ++k) {
    float di = dv[k][n];
    long o = (long)n * CW + k * M + f;
    float v = xcat_in[o] + di * di * hv + bv;
    v = v > 0.0f ? v : 0.0f;
    xout[o] = v;
    xbf[o]  = f2bf(v);
  }
}

// ---------------------------------------------------------------------------
// Final Conv1d(k=1) 384->4 + log_softmax, one thread per node.
// ---------------------------------------------------------------------------
__global__ void kFinal(const float* __restrict__ x, const float* __restrict__ cw,
                       const float* __restrict__ cb, float* __restrict__ out,
                       int N, int D) {
  int n = blockIdx.x * blockDim.x + threadIdx.x;
  if (n >= N) return;
  float acc[4] = { cb[0], cb[1], cb[2], cb[3] };
  const float* xr = x + (long)n * D;
  for (int t = 0; t < D; ++t) {
    float xv = xr[t];
    #pragma unroll
    for (int j = 0; j < 4; ++j) acc[j] += xv * cw[j * D + t];
  }
  float mx = fmaxf(fmaxf(acc[0], acc[1]), fmaxf(acc[2], acc[3]));
  float s = 0.0f;
  #pragma unroll
  for (int j = 0; j < 4; ++j) s += __expf(acc[j] - mx);
  float ls = __logf(s);
  #pragma unroll
  for (int j = 0; j < 4; ++j) out[(long)n * 4 + j] = acc[j] - mx - ls;
}

// ---------------------------------------------------------------------------
// Host orchestration
// ---------------------------------------------------------------------------
extern "C" void kernel_launch(void* const* d_in, const int* in_sizes, int n_in,
                              void* d_out, int out_size, void* d_ws, size_t ws_size,
                              hipStream_t stream) {
  const float* x_in  = (const float*)d_in[0];
  const int*   eidx  = (const int*)d_in[1];
  const int*   ein   = (const int*)d_in[2];
  const int*   eout  = (const int*)d_in[3];
  const float* in_w  = (const float*)d_in[4];
  const float* out_w = (const float*)d_in[5];
  const float* W0 = (const float*)d_in[6];
  const float* W1 = (const float*)d_in[7];
  const float* W2 = (const float*)d_in[8];
  const float* b0 = (const float*)d_in[9];
  const float* b1 = (const float*)d_in[10];
  const float* b2 = (const float*)d_in[11];
  const float* cw = (const float*)d_in[12];
  const float* cb = (const float*)d_in[13];

  const int N = in_sizes[0] / 128;        // 50000
  const int Ea = in_sizes[1] / 2;
  const int Eb = in_sizes[2] / 2;
  const int Ec = in_sizes[3] / 2;

  // ---- workspace layout ----
  char* ws = (char*)d_ws;
  size_t off = 0;
  auto take = [&](size_t bytes) -> char* {
    char* p = ws + off;
    off = (off + bytes + 255) & ~(size_t)255;
    return p;
  };
  float* dinvA = (float*)take((size_t)N * 4);
  float* dinvB = (float*)take((size_t)N * 4);
  float* dinvC = (float*)take((size_t)N * 4);
  float* xcat  = (float*)take((size_t)N * 768 * 4);          // concat/scatter accumulator
  unsigned short* xbf = (unsigned short*)take((size_t)N * 768 * 2);  // bf16 activations
  float* h     = (float*)take((size_t)N * 256 * 4);          // GEMM output
  unsigned short* wt = (unsigned short*)take((size_t)768 * 256 * 2); // bf16 W^T

  const int T = 256;

  // ---- dinv for the three edge sets (deg = 1 (self loop) + sum of weights) ----
  const int*   srcs[3] = { eidx,       ein,       eout      };
  const int*   dsts[3] = { eidx + Ea,  ein + Eb,  eout + Ec };
  const float* ews[3]  = { nullptr,    in_w,      out_w     };
  const int    Es[3]   = { Ea,         Eb,        Ec        };
  float*       dinvs[3] = { dinvA, dinvB, dinvC };

  for (int k = 0; k < 3; ++k) {
    kFillOnes<<<(N + T - 1) / T, T, 0, stream>>>(dinvs[k], N);
    kDegAccum<<<(Es[k] + T - 1) / T, T, 0, stream>>>(dsts[k], ews[k], dinvs[k], Es[k]);
    kRsqrt<<<(N + T - 1) / T, T, 0, stream>>>(dinvs[k], N);
  }

  // ---- initial bf16 activations ----
  kConvertBf<<<(int)(((long)N * 128 + T - 1) / T), T, 0, stream>>>(x_in, xbf, (long)N * 128);

  // ---- three DGCN layers ----
  struct Layer { const float* W; const float* b; int K, M; };
  Layer L[3] = { { W0, b0, 128, 256 }, { W1, b1, 768, 256 }, { W2, b2, 768, 128 } };

  for (int l = 0; l < 3; ++l) {
    const int K = L[l].K, M = L[l].M, CW = 3 * M;

    // W^T -> bf16
    kTransposeConvertW<<<(K * M + T - 1) / T, T, 0, stream>>>(L[l].W, wt, K, M);

    // h[N, M] = xbf[N, K] @ W[K, M]   (WMMA bf16, f32 accumulate, TDM-staged A)
    int bmCount = (N + 15) / 16;
    int blocks = bmCount * (M >> 7);
    kGemmWmmaBf16<<<blocks, T, 0, stream>>>(xbf, wt, h, N, K, M);

    // zero concat accumulator, scatter 3 convs into it
    kZero<<<2048, T, 0, stream>>>(xcat, (long)N * CW);
    const int EPB = 16;
    for (int k = 0; k < 3; ++k) {
      kScatter<<<(Es[k] + EPB - 1) / EPB, T, 0, stream>>>(
          srcs[k], dsts[k], ews[k], dinvs[k], h, xcat, Es[k], M, CW, k * M, EPB);
    }

    // self-loop + bias + ReLU; produce fp32 (into xcat, or d_out for last layer) + bf16
    float* xout = (l == 2) ? (float*)d_out : xcat;
    kEpilogue<<<(N * M + T - 1) / T, T, 0, stream>>>(
        xcat, xout, xbf, h, dinvA, dinvB, dinvC, L[l].b, N, M);
  }

  // ---- final linear 384->4 + log_softmax; x already sits at d_out[0 .. N*384) ----
  kFinal<<<(N + T - 1) / T, T, 0, stream>>>(
      (const float*)d_out, cw, cb, (float*)d_out + (long)N * 384, N, 384);
}
